// EquivariantGNN_52080773431365
// MI455X (gfx1250) — matrix operand order repaired and checked by reading, not compiled, not run on hardware
//
#include <hip/hip_runtime.h>
#include <hip/hip_bf16.h>
#include <math.h>

// ---------------------------------------------------------------------------
// Equivariant GNN forward for MI455X (gfx1250, wave32, WMMA).
// E x 64 x 4800 radial-weight GEMM fused with the tensor product (tpw never
// materialized). v_wmma_f32_16x16x32_f16 per 16-col subtile; path-specialized
// templates fold all index math; B operands software-pipelined; y gathered in
// bulk to registers so ds_add_f32 streams without per-element dscnt waits.
// ---------------------------------------------------------------------------

typedef __attribute__((ext_vector_type(16))) _Float16 v16h;
typedef __attribute__((ext_vector_type(8)))  _Float16 v8h;
typedef __attribute__((ext_vector_type(8)))  float    v8f;

// ---- path metadata (compile-time, used by templates) ----------------------
constexpr int K_I1[17]  = {0,0,0,0,0,1,1,1,1,1,1,2,2,2,2,2,2};
constexpr int K_I2[17]  = {0,0,0,1,2,0,1,1,1,1,2,0,1,2,2,2,2};
constexpr int K_IO[17]  = {0,1,2,3,4,3,0,1,2,4,3,4,3,0,1,2,4};
constexpr int K_WS[17]  = {0,1024,1536,1792,2304,2560,2816,3328,
                           3584,3712,3840,4096,4160,4288,4544,4672,4736};
constexpr int K_MULS[3] = {32,16,8};
constexpr int K_HOFF[3] = {0,32,80};
constexpr int K_SHOFF[3]= {0,1,4};
constexpr int K_TPMUL[5]= {32,16,8,16,8};
constexpr int K_TPL[5]  = {0,0,0,1,2};
constexpr int K_OUTOFF[5]={0,32,48,56,104};
// coeff = sqrt((2*l3+1)/fan): fan = 56,56,56,72,64
constexpr float K_COEFF[17] = {
  0.13363062095621219f,0.13363062095621219f,0.13363062095621219f,
  0.20412414523193154f,0.27950849718747373f,0.20412414523193154f,
  0.13363062095621219f,0.13363062095621219f,0.13363062095621219f,
  0.27950849718747373f,0.20412414523193154f,0.27950849718747373f,
  0.20412414523193154f,0.13363062095621219f,0.13363062095621219f,
  0.13363062095621219f,0.27950849718747373f};

// ---- runtime-indexed copies for the w3j init kernel -----------------------
__device__ __constant__ int c_P_I1[17] = {0,0,0,0,0,1,1,1,1,1,1,2,2,2,2,2,2};
__device__ __constant__ int c_P_I2[17] = {0,0,0,1,2,0,1,1,1,1,2,0,1,2,2,2,2};
__device__ __constant__ int c_P_IO[17] = {0,1,2,3,4,3,0,1,2,4,3,4,3,0,1,2,4};
__device__ __constant__ int c_TPL[5]   = {0,0,0,1,2};

// ---------------------------------------------------------------------------
// Kernel 1: Wigner-3j change-of-basis tensors (exact port of _cg/_real_u/_w3j)
// ---------------------------------------------------------------------------
__device__ double dfac(int n){ double r=1.0; for(int i=2;i<=n;++i) r*=(double)i; return r; }
__device__ int imax3(int a,int b,int c){ int m=a; if(b>m)m=b; if(c>m)m=c; return m; }
__device__ int imin3(int a,int b,int c){ int m=a; if(b<m)m=b; if(c<m)m=c; return m; }

__device__ double cg_coef(int l1,int l2,int l3,int m1,int m2,int m3){
  if (m1+m2 != m3) return 0.0;
  double pre = sqrt((2.0*l3+1.0)*dfac(l3+l1-l2)*dfac(l3-l1+l2)*dfac(l1+l2-l3)/dfac(l1+l2+l3+1));
  pre *= sqrt(dfac(l3+m3)*dfac(l3-m3)*dfac(l1-m1)*dfac(l1+m1)*dfac(l2-m2)*dfac(l2+m2));
  int k0 = imax3(0, l2-l3-m1, l1-l3+m2);
  int k1 = imin3(l1+l2-l3, l1-m1, l2+m2);
  double s=0.0;
  for(int k=k0;k<=k1;++k){
    double den = dfac(k)*dfac(l1+l2-l3-k)*dfac(l1-m1-k)*dfac(l2+m2-k)*dfac(l3-l2+m1+k)*dfac(l3-l1-m2+k);
    s += ((k&1)?-1.0:1.0)/den;
  }
  return pre*s;
}

__device__ void real_u(int l, double ur[5][5], double ui[5][5]){
  for(int a=0;a<5;a++) for(int b=0;b<5;b++){ ur[a][b]=0.0; ui[a][b]=0.0; }
  ur[l][l]=1.0;
  double is2 = 1.0/sqrt(2.0);
  for(int m=1;m<=l;++m){
    double sgn = (m&1)?-1.0:1.0;
    ur[l+m][l+m] = sgn*is2;
    ur[l+m][l-m] = is2;
    ui[l-m][l-m] = is2;
    ui[l-m][l+m] = -sgn*is2;
  }
}

__global__ void w3j_init_kernel(float* __restrict__ Cbuf){
  int p = threadIdx.x;
  if (p >= 17) return;
  int l1 = c_P_I1[p], l2 = c_P_I2[p], l3 = c_TPL[c_P_IO[p]];
  int K1=2*l1+1, K2=2*l2+1, K3=2*l3+1;
  double u1r[5][5],u1i[5][5],u2r[5][5],u2i[5][5],u3r[5][5],u3i[5][5];
  real_u(l1,u1r,u1i); real_u(l2,u2r,u2i); real_u(l3,u3r,u3i);
  double Tr[125], Ti[125];
  double mr=0.0, mi=0.0;
  for(int a=0;a<K1;a++) for(int b=0;b<K2;b++) for(int c=0;c<K3;c++){
    double tr=0.0, ti=0.0;
    for(int m=0;m<K1;m++) for(int n=0;n<K2;n++) for(int q=0;q<K3;q++){
      double cgv = cg_coef(l1,l2,l3, m-l1, n-l2, q-l3);
      if (cgv==0.0) continue;
      double ar=u1r[a][m], ai=u1i[a][m];
      double br=u2r[b][n], bi=u2i[b][n];
      double cr=u3r[c][q], ci=-u3i[c][q];     // conj
      double abr = ar*br - ai*bi, abi = ar*bi + ai*br;
      double zr  = abr*cr - abi*ci, zi = abr*ci + abi*cr;
      tr += zr*cgv; ti += zi*cgv;
    }
    Tr[(a*K2+b)*K3+c]=tr; Ti[(a*K2+b)*K3+c]=ti;
    if (fabs(tr)>mr) mr=fabs(tr);
    if (fabs(ti)>mi) mi=fabs(ti);
  }
  double inv = 1.0/sqrt((double)(2*l3+1));
  int tot = K1*K2*K3;
  for(int i=0;i<tot;++i)
    Cbuf[p*128+i] = (float)(((mr>=mi)?Tr[i]:Ti[i])*inv);
}

// ---------------------------------------------------------------------------
// Kernel 2: node init: species argmax -> embed -> proj / sqrt(32); zero rest
// ---------------------------------------------------------------------------
__global__ void node_init_kernel(const float* __restrict__ feats,
                                 const float* __restrict__ embed,
                                 const float* __restrict__ projw,
                                 float* __restrict__ h){
  int n = blockIdx.x, lane = threadIdx.x;
  float best=-1e30f; int bi=0;
  for(int c=lane;c<86;c+=32){ float v=feats[(size_t)n*86+c]; if(v>best){best=v;bi=c;} }
  for(int off=16; off; off>>=1){
    float ov = __shfl_down(best, off);
    int   oi = __shfl_down(bi,   off);
    if (ov>best || (ov==best && oi<bi)){ best=ov; bi=oi; }
  }
  int sp = __shfl(bi, 0);
  float acc=0.f;
  for(int k=0;k<32;k++) acc += embed[sp*32+k]*projw[k*32+lane];
  h[(size_t)n*120+lane] = acc*0.17677669529663687f;   // 1/sqrt(32)
  for(int c=32+lane;c<120;c+=32) h[(size_t)n*120+c]=0.f;
}

// ---------------------------------------------------------------------------
// Kernel 3: edge geometry: spherical harmonics + smoothed radial basis
// ---------------------------------------------------------------------------
__global__ void edge_geom_kernel(const float* __restrict__ vec,
                                 const float* __restrict__ freqs,
                                 float* __restrict__ sh,
                                 float* __restrict__ basis, int E){
  int e = blockIdx.x*blockDim.x + threadIdx.x; if(e>=E) return;
  float x=vec[e*3], y=vec[e*3+1], z=vec[e*3+2];
  float d = sqrtf(x*x+y*y+z*z);
  float ir = 1.f/fmaxf(d,1e-9f);
  float ux=x*ir, uy=y*ir, uz=z*ir;
  const float s3=1.7320508075688772f, s15=3.872983346207417f, s5=2.23606797749979f;
  float* S = sh+(size_t)e*9;
  S[0]=1.f; S[1]=s3*uy; S[2]=s3*uz; S[3]=s3*ux;
  S[4]=s15*ux*uy; S[5]=s15*uy*uz; S[6]=0.5f*s5*(3.f*uz*uz-1.f);
  S[7]=s15*ux*uz; S[8]=0.5f*s15*(ux*ux-uy*uy);
  float xr=d*0.2f, xr2=xr*xr, xr3=xr2*xr;
  float cut = 1.f - 6.f*xr3*xr2 + 15.f*xr2*xr2 - 10.f*xr3;
  float idd = 1.f/fmaxf(d,1e-6f);
  for(int b=0;b<8;b++) basis[(size_t)e*8+b] = sinf(freqs[b]*d)*idd*cut;
}

// ---------------------------------------------------------------------------
// Kernel 4 (per layer): radial MLP layers 1+2 (tiny) -> z2 in f16
// ---------------------------------------------------------------------------
__global__ void radial12_kernel(const float* __restrict__ basis,
                                const float* __restrict__ w1, const float* __restrict__ b1,
                                const float* __restrict__ w2, const float* __restrict__ b2,
                                _Float16* __restrict__ z2, int E){
  __shared__ float sW1[8*64], sB1[64], sW2[64*64], sB2[64];
  int tid=threadIdx.x;
  for(int i=tid;i<512;i+=64)  sW1[i]=w1[i];
  if (tid<64){ sB1[tid]=b1[tid]; sB2[tid]=b2[tid]; }
  for(int i=tid;i<4096;i+=64) sW2[i]=w2[i];
  __syncthreads();
  int e = blockIdx.x*64+tid; if(e>=E) return;
  float bb[8];
  #pragma unroll
  for(int i=0;i<8;i++) bb[i]=basis[(size_t)e*8+i];
  float z1[64];
  for(int j=0;j<64;j++){
    float a=sB1[j];
    #pragma unroll
    for(int i=0;i<8;i++) a += bb[i]*sW1[i*64+j];
    z1[j] = a/(1.f+__expf(-a));          // silu
  }
  for(int j=0;j<64;j++){
    float a=sB2[j];
    for(int i=0;i<64;i++) a += z1[i]*sW2[i*64+j];
    float s = a/(1.f+__expf(-a));
    z2[(size_t)e*64+j] = (_Float16)s;
  }
}

// ---------------------------------------------------------------------------
// Kernel 5 (per layer): pack w3 (64 x 4800, row-major) -> [col][k] f16
// ---------------------------------------------------------------------------
__global__ void pack_w3_kernel(const float* __restrict__ w3, _Float16* __restrict__ pack){
  int idx = blockIdx.x*256+threadIdx.x;           // < 4800*64
  int col = idx>>6, k = idx&63;
  pack[idx] = (_Float16)w3[(size_t)k*4800+col];
}

__global__ void zero_kernel(float* __restrict__ p, int n){
  int i = blockIdx.x*256+threadIdx.x; if(i<n) p[i]=0.f;
}

// ---------------------------------------------------------------------------
// Helper: 2x WMMA K=64 accumulate
// ---------------------------------------------------------------------------
__device__ __forceinline__ v8f wmma64(const v16h& a0, const v16h& a1,
                                      const v16h& b0, const v16h& b1){
  v8f acc = {};
  acc = __builtin_amdgcn_wmma_f32_16x16x32_f16(false, a0, false, b0, (short)0, acc, false, false);
  acc = __builtin_amdgcn_wmma_f32_16x16x32_f16(false, a1, false, b1, (short)0, acc, false, false);
  return acc;
}

// ---------------------------------------------------------------------------
// Path-specialized tensor-product step. All metadata compile-time.
//  * y[e,u,k] via unrolled small contractions (no division anywhere)
//  * MULW==32 paths (K3==1): subtile PAIRS share u -> one y gather feeds two
//    16-col tiles; 4 WMMAs in flight per iteration.
//  * y gathered in bulk to registers between the WMMAs and the acc reads:
//    fills the WMMA->VALU hazard window and lets ds_add_f32 stream w/o waits.
// ---------------------------------------------------------------------------
template<int P>
__device__ __forceinline__ void tp_path(
    int lane, const v16h& a0, const v16h& a1,
    const _Float16* __restrict__ w3p, const float* __restrict__ b3,
    const float* xs, const float* shs, const float* Cl,
    float* msg, float* ybuf)
{
  constexpr int i1   = K_I1[P], i2 = K_I2[P], io = K_IO[P];
  constexpr int l3   = K_TPL[io];
  constexpr int K1   = 2*i1+1, K2 = 2*i2+1, K3 = 2*l3+1;
  constexpr int MUL1 = K_MULS[i1], MULW = K_TPMUL[io];
  constexpr int HOFF = K_HOFF[i1], SHOFF = K_SHOFF[i2], OUTOFF = K_OUTOFF[io];
  constexpr int WS   = K_WS[P];
  constexpr int NSUB = (MUL1*MULW)/16;
  constexpr int LOGW = (MULW==32)?5:((MULW==16)?4:3);
  const float coeff  = K_COEFF[P];

  // ---- y[e][u][k]: two lanes per edge row, strided over u -----------------
  {
    int m = lane>>1, sub = lane&1;
    const float* xp0 = &xs[m*120 + HOFF];
    float shv[K2];
    #pragma unroll
    for(int j=0;j<K2;j++) shv[j] = shs[m*9 + SHOFF + j];
    float* yb = &ybuf[m*MUL1*K3];
    for(int u=sub; u<MUL1; u+=2){
      const float* xp = xp0 + u*K1;
      float acc[K3];
      #pragma unroll
      for(int k=0;k<K3;k++) acc[k]=0.f;
      #pragma unroll
      for(int i=0;i<K1;i++){
        float xv = xp[i];
        #pragma unroll
        for(int j=0;j<K2;j++){
          float t = xv*shv[j];
          #pragma unroll
          for(int k=0;k<K3;k++) acc[k] += t * Cl[P*128 + (i*K2+j)*K3 + k];
        }
      }
      #pragma unroll
      for(int k=0;k<K3;k++) yb[u*K3+k]=acc[k];
    }
  }
  __syncthreads();

  const int lc = lane&15, halfId = lane>>4;
  const int mbase = halfId*8;
  const _Float16* bbase = w3p + (size_t)(WS+lc)*64 + halfId*16;

  if constexpr (MULW==32){
    // ---- paired subtiles: u = pair index (uniform), K3 == 1 ---------------
    v8h qa0=*(const v8h*)(bbase),      qa1=*(const v8h*)(bbase+8);
    v8h qa2=*(const v8h*)(bbase+32),   qa3=*(const v8h*)(bbase+40);
    const _Float16* bb2 = bbase + 16*64;
    v8h qb0=*(const v8h*)(bb2),        qb1=*(const v8h*)(bb2+8);
    v8h qb2=*(const v8h*)(bb2+32),     qb3=*(const v8h*)(bb2+40);
    float biasA = b3[WS+lc], biasB = b3[WS+16+lc];
    for(int cp=0; cp<NSUB/2; ++cp){
      v16h b0,b1,c0,c1;
      #pragma unroll
      for(int t=0;t<8;t++){
        b0[t]=qa0[t]; b0[t+8]=qa1[t]; b1[t]=qa2[t]; b1[t+8]=qa3[t];
        c0[t]=qb0[t]; c0[t+8]=qb1[t]; c1[t]=qb2[t]; c1[t+8]=qb3[t];
      }
      v8f accA = wmma64(a0,a1,b0,b1);
      v8f accB = wmma64(a0,a1,c0,c1);
      float cbA = coeff*biasA, cbB = coeff*biasB;
      if (cp+1 < NSUB/2){                       // preload next pair
        const _Float16* bpA = bbase + (size_t)(cp+1)*32*64;
        const _Float16* bpB = bpA + 16*64;
        qa0=*(const v8h*)(bpA);    qa1=*(const v8h*)(bpA+8);
        qa2=*(const v8h*)(bpA+32); qa3=*(const v8h*)(bpA+40);
        qb0=*(const v8h*)(bpB);    qb1=*(const v8h*)(bpB+8);
        qb2=*(const v8h*)(bpB+32); qb3=*(const v8h*)(bpB+40);
        biasA = b3[WS+(cp+1)*32+lc]; biasB = b3[WS+(cp+1)*32+16+lc];
      }
      const int u = cp;
      float yv[8];
      #pragma unroll
      for(int r=0;r<8;r++) yv[r] = ybuf[(r+mbase)*MUL1+u];
      #pragma unroll
      for(int r=0;r<8;r++){
        float* mp = &msg[(r+mbase)*144 + OUTOFF];
        atomicAdd(&mp[lc],    fmaf(coeff, accA[r], cbA)*yv[r]);
        atomicAdd(&mp[16+lc], fmaf(coeff, accB[r], cbB)*yv[r]);
      }
    }
  } else {
    // ---- generic path (MULW == 16 or 8) -----------------------------------
    v8h q0=*(const v8h*)(bbase),    q1=*(const v8h*)(bbase+8);
    v8h q2=*(const v8h*)(bbase+32), q3=*(const v8h*)(bbase+40);
    float bias = b3[WS+lc];
    for(int ct=0; ct<NSUB; ++ct){
      v16h b0,b1;
      #pragma unroll
      for(int t=0;t<8;t++){ b0[t]=q0[t]; b0[t+8]=q1[t]; b1[t]=q2[t]; b1[t+8]=q3[t]; }
      v8f acc = wmma64(a0,a1,b0,b1);
      float cb = coeff*bias;
      int pcol = ct*16 + lc;
      if (ct+1 < NSUB){                         // preload next subtile
        const _Float16* bp = bbase + (size_t)(ct+1)*16*64;
        q0=*(const v8h*)(bp);    q1=*(const v8h*)(bp+8);
        q2=*(const v8h*)(bp+32); q3=*(const v8h*)(bp+40);
        bias = b3[WS+(ct+1)*16+lc];
      }
      const int u = pcol>>LOGW, w = pcol&(MULW-1);
      float yv[8*K3];
      #pragma unroll
      for(int r=0;r<8;r++)
        #pragma unroll
        for(int k=0;k<K3;k++) yv[r*K3+k] = ybuf[((r+mbase)*MUL1+u)*K3+k];
      #pragma unroll
      for(int r=0;r<8;r++){
        float cw = fmaf(coeff, acc[r], cb);
        float* mp = &msg[(r+mbase)*144 + OUTOFF + w*K3];
        #pragma unroll
        for(int k=0;k<K3;k++) atomicAdd(&mp[k], cw*yv[r*K3+k]);
      }
    }
  }
  __syncthreads();
}

// ---------------------------------------------------------------------------
// Kernel 6 (per layer): FUSED radial-w3 GEMM (WMMA) + tensor product + scatter
// One wave (32 lanes) per 16-edge tile.
// ---------------------------------------------------------------------------
__global__ void __launch_bounds__(32) fused_tp_kernel(
    const _Float16* __restrict__ z2, const _Float16* __restrict__ w3p,
    const float* __restrict__ b3,   const float* __restrict__ h,
    const float* __restrict__ sh,   const int* __restrict__ eidx,
    const float* __restrict__ Cbuf, float* __restrict__ agg, int E)
{
  __shared__ float xs[16*120];       // gathered source features
  __shared__ float shs[16*9];        // spherical harmonics
  __shared__ float msg[16*144];      // per-tile message accumulator
  __shared__ float ybuf[16*32*5];    // y[e][u][k] for current path
  __shared__ float Cl[17*128];       // all w3j tensors
  __shared__ int   ssrc[16], sdst[16];

  int lane = threadIdx.x;
  int et   = blockIdx.x*16;
  if (lane<16){ ssrc[lane]=eidx[et+lane]; sdst[lane]=eidx[E+et+lane]; }
  for(int i=lane;i<17*128;i+=32) Cl[i]=Cbuf[i];
  for(int i=lane;i<16*144;i+=32) msg[i]=0.f;
  for(int i=lane;i<16*9;  i+=32) shs[i]=sh[(size_t)et*9+i];
  __syncthreads();
  // vectorized gather of 16 source rows (120 floats = 30 float4 each)
  for(int i=lane;i<480;i+=32){
    int m=i/30, o=(i-m*30)*4;
    float4 v = *(const float4*)(h + (size_t)ssrc[m]*120 + o);
    *(float4*)&xs[m*120+o] = v;
  }

  // --- A operand: z2 rows in WMMA 16-bit A layout ---------------------------
  int halfId = lane>>4;              // 0: K{0..15}  1: K{16..31}
  int mrow   = lane & 15;            // M row
  const _Float16* zr = z2 + (size_t)(et+mrow)*64 + halfId*16;
  v8h p0 = *(const v8h*)(zr);
  v8h p1 = *(const v8h*)(zr+8);
  v8h p2 = *(const v8h*)(zr+32);
  v8h p3 = *(const v8h*)(zr+40);
  v16h a0, a1;
  #pragma unroll
  for(int t=0;t<8;t++){ a0[t]=p0[t]; a0[t+8]=p1[t]; a1[t]=p2[t]; a1[t+8]=p3[t]; }
  if (et+16 < E) __builtin_prefetch(z2 + (size_t)(et+16+mrow)*64, 0, 1);
  __syncthreads();

  tp_path< 0>(lane,a0,a1,w3p,b3,xs,shs,Cl,msg,ybuf);
  tp_path< 1>(lane,a0,a1,w3p,b3,xs,shs,Cl,msg,ybuf);
  tp_path< 2>(lane,a0,a1,w3p,b3,xs,shs,Cl,msg,ybuf);
  tp_path< 3>(lane,a0,a1,w3p,b3,xs,shs,Cl,msg,ybuf);
  tp_path< 4>(lane,a0,a1,w3p,b3,xs,shs,Cl,msg,ybuf);
  tp_path< 5>(lane,a0,a1,w3p,b3,xs,shs,Cl,msg,ybuf);
  tp_path< 6>(lane,a0,a1,w3p,b3,xs,shs,Cl,msg,ybuf);
  tp_path< 7>(lane,a0,a1,w3p,b3,xs,shs,Cl,msg,ybuf);
  tp_path< 8>(lane,a0,a1,w3p,b3,xs,shs,Cl,msg,ybuf);
  tp_path< 9>(lane,a0,a1,w3p,b3,xs,shs,Cl,msg,ybuf);
  tp_path<10>(lane,a0,a1,w3p,b3,xs,shs,Cl,msg,ybuf);
  tp_path<11>(lane,a0,a1,w3p,b3,xs,shs,Cl,msg,ybuf);
  tp_path<12>(lane,a0,a1,w3p,b3,xs,shs,Cl,msg,ybuf);
  tp_path<13>(lane,a0,a1,w3p,b3,xs,shs,Cl,msg,ybuf);
  tp_path<14>(lane,a0,a1,w3p,b3,xs,shs,Cl,msg,ybuf);
  tp_path<15>(lane,a0,a1,w3p,b3,xs,shs,Cl,msg,ybuf);
  tp_path<16>(lane,a0,a1,w3p,b3,xs,shs,Cl,msg,ybuf);

  // scatter message tile to agg
  for(int i=lane;i<16*144;i+=32){
    int m=i/144;
    atomicAdd(&agg[(size_t)sdst[m]*144 + (i - m*144)], msg[i]);
  }
}

// ---------------------------------------------------------------------------
// Kernel 7 (per layer): gating + self-interaction + global norm stats
// ---------------------------------------------------------------------------
__global__ void node_update_kernel(const float* __restrict__ agg,
                                   const float* __restrict__ w0,
                                   const float* __restrict__ w1s,
                                   const float* __restrict__ w2s,
                                   float* __restrict__ tmp, float* __restrict__ stats,
                                   float csilu, float csig){
  __shared__ float sS[32], sG[24], sV1[48], sV2[40];
  int n = blockIdx.x, lane = threadIdx.x;
  const float* a = agg + (size_t)n*144;
  { float v=a[lane]; sS[lane] = csilu * v/(1.f+__expf(-v)); }
  if (lane<24){ float v=a[32+lane]; sG[lane] = csig/(1.f+__expf(-v)); }
  __syncthreads();
  for(int i=lane;i<48;i+=32){ int u=i/3; sV1[i] = a[56+i]*sG[u]; }
  for(int i=lane;i<40;i+=32){ int u=i/5; sV2[i] = a[104+i]*sG[16+u]; }
  __syncthreads();
  float accs=0.f;
  for(int u=0;u<32;u++) accs += sS[u]*w0[u*32+lane];
  accs *= 0.17677669529663687f;                       // /sqrt(32)
  tmp[(size_t)n*120+lane]=accs;
  atomicAdd(&stats[lane], accs);
  atomicAdd(&stats[32+lane], accs*accs);
  for(int i=lane;i<48;i+=32){
    int v=i/3, dd=i-v*3;
    float acc=0.f;
    for(int u=0;u<16;u++) acc += sV1[u*3+dd]*w1s[u*16+v];
    acc *= 0.25f;                                     // /sqrt(16)
    tmp[(size_t)n*120+32+i]=acc;
    atomicAdd(&stats[64+v], acc*acc);
  }
  for(int i=lane;i<40;i+=32){
    int v=i/5, dd=i-v*5;
    float acc=0.f;
    for(int u=0;u<8;u++) acc += sV2[u*5+dd]*w2s[u*8+v];
    acc *= 0.35355339059327373f;                      // /sqrt(8)
    tmp[(size_t)n*120+80+i]=acc;
    atomicAdd(&stats[80+v], acc*acc);
  }
}

__global__ void scales_kernel(const float* __restrict__ stats,
                              const float* __restrict__ bw,
                              float* __restrict__ scales, int N){
  int j=threadIdx.x;
  float invN = 1.f/(float)N;
  if(j<32){
    float mean = stats[j]*invN;
    float var  = stats[32+j]*invN - mean*mean;
    scales[j]    = mean;
    scales[32+j] = rsqrtf(var+1e-5f)*bw[j];
  } else if (j<48){
    int v=j-32;
    scales[64+v] = rsqrtf(stats[64+v]*invN*(1.f/3.f)+1e-5f)*bw[32+v];
  } else if (j<56){
    int v=j-48;
    scales[80+v] = rsqrtf(stats[80+v]*invN*0.2f+1e-5f)*bw[48+v];
  }
}

__global__ void apply_norm_kernel(const float* __restrict__ tmp,
                                  const float* __restrict__ scales,
                                  const float* __restrict__ bb,
                                  float* __restrict__ h, int N){
  int idx = blockIdx.x*256+threadIdx.x; if(idx>=N*120) return;
  int n=idx/120, c=idx-n*120; (void)n;
  float t=tmp[idx], o;
  if(c<32)      o = (t - scales[c])*scales[32+c] + bb[c];
  else if(c<80) o = t*scales[64 + (c-32)/3];
  else          o = t*scales[80 + (c-80)/5];
  h[idx] += o;
}

// ---------------------------------------------------------------------------
// Pooling + head MLP
// ---------------------------------------------------------------------------
__global__ void pool_kernel(const float* __restrict__ h, const int* __restrict__ batch,
                            float* __restrict__ gsum, int N){
  int n = blockIdx.x*blockDim.x+threadIdx.x; if(n>=N) return;
  int b=batch[n];
  atomicAdd(&gsum[512+b], 1.f);
  for(int c=0;c<32;c++) atomicAdd(&gsum[b*32+c], h[(size_t)n*120+c]);
}

__global__ void head_kernel(const float* __restrict__ gsum,
                            const float* __restrict__ w1,const float* __restrict__ bb1,
                            const float* __restrict__ w2,const float* __restrict__ bb2,
                            const float* __restrict__ w3,const float* __restrict__ bb3,
                            float* __restrict__ out){
  int g=threadIdx.x; if(g>=16) return;
  float cnt = fmaxf(gsum[512+g],1.f);
  float emb[32];
  for(int c=0;c<32;c++) emb[c]=gsum[g*32+c]/cnt;
  float z1[32];
  for(int j=0;j<32;j++){ float a=bb1[j]; for(int k=0;k<32;k++) a+=emb[k]*w1[k*32+j]; z1[j]=a/(1.f+__expf(-a)); }
  float z2v[16];
  for(int j=0;j<16;j++){ float a=bb2[j]; for(int k=0;k<32;k++) a+=z1[k]*w2[k*16+j]; z2v[j]=a/(1.f+__expf(-a)); }
  float a=bb3[0];
  for(int k=0;k<16;k++) a+=z2v[k]*w3[k];
  out[g]=a;
}

// ---------------------------------------------------------------------------
// Host launcher
// ---------------------------------------------------------------------------
extern "C" void kernel_launch(void* const* d_in, const int* in_sizes, int n_in,
                              void* d_out, int out_size, void* d_ws, size_t ws_size,
                              hipStream_t stream) {
  (void)n_in; (void)out_size; (void)ws_size;
  const float* node_feats = (const float*)d_in[0];
  const int*   eidx       = (const int*)  d_in[1];
  const float* evec       = (const float*)d_in[2];
  const int*   batch      = (const int*)  d_in[3];
  const float* embed      = (const float*)d_in[4];
  const float* projw      = (const float*)d_in[5];
  const float* freqs      = (const float*)d_in[6];
  const float* w1         = (const float*)d_in[7];
  const float* b1         = (const float*)d_in[8];
  const float* w2         = (const float*)d_in[9];
  const float* b2         = (const float*)d_in[10];
  const float* w3         = (const float*)d_in[11];
  const float* b3         = (const float*)d_in[12];
  const float* si0        = (const float*)d_in[13];
  const float* si1        = (const float*)d_in[14];
  const float* si2        = (const float*)d_in[15];
  const float* bnw        = (const float*)d_in[16];
  const float* bnb        = (const float*)d_in[17];
  const float* hw1        = (const float*)d_in[18];
  const float* hb1        = (const float*)d_in[19];
  const float* hw2        = (const float*)d_in[20];
  const float* hb2        = (const float*)d_in[21];
  const float* hw3        = (const float*)d_in[22];
  const float* hb3        = (const float*)d_in[23];

  const int N = in_sizes[0]/86;
  const int E = in_sizes[2]/3;

  // workspace carve (256B aligned)
  char* cur = (char*)d_ws;
  auto carve = [&](size_t bytes)->char*{
    char* r = cur; cur += (bytes + 255) & ~((size_t)255); return r;
  };
  float*    Cbuf   = (float*)   carve(17*128*sizeof(float));
  float*    h      = (float*)   carve((size_t)N*120*sizeof(float));
  float*    tmp    = (float*)   carve((size_t)N*120*sizeof(float));
  float*    shb    = (float*)   carve((size_t)E*9*sizeof(float));
  float*    basis  = (float*)   carve((size_t)E*8*sizeof(float));
  _Float16* z2     = (_Float16*)carve((size_t)E*64*sizeof(_Float16));
  _Float16* w3p    = (_Float16*)carve((size_t)4800*64*sizeof(_Float16));
  float*    agg    = (float*)   carve((size_t)N*144*sizeof(float));
  float*    stats  = (float*)   carve(128*sizeof(float));
  float*    scales = (float*)   carve(128*sizeof(float));
  float*    gsum   = (float*)   carve(528*sizeof(float));

  // host-side normalization constants (exact port of reference integrals)
  double ssi=0.0, ssg=0.0;
  const double TWO_PI = 6.283185307179586476925286766559;
  for(int i=0;i<8001;i++){
    double z = -8.0 + 16.0*(double)i/8000.0;
    double phi = exp(-z*z/2.0)/sqrt(TWO_PI);
    double sil = z/(1.0+exp(-z));
    double sig = 1.0/(1.0+exp(-z));
    ssi += sil*sil*phi; ssg += sig*sig*phi;
  }
  double dz = 16.0/8000.0;
  float csilu = (float)(1.0/sqrt(ssi*dz));
  float csig  = (float)(1.0/sqrt(ssg*dz));

  // --- setup ---
  w3j_init_kernel<<<1, 32, 0, stream>>>(Cbuf);
  node_init_kernel<<<N, 32, 0, stream>>>(node_feats, embed, projw, h);
  edge_geom_kernel<<<(E+255)/256, 256, 0, stream>>>(evec, freqs, shb, basis, E);

  // --- layers ---
  for(int l=0;l<3;l++){
    radial12_kernel<<<(E+63)/64, 64, 0, stream>>>(
        basis, w1 + (size_t)l*512, b1 + (size_t)l*64,
        w2 + (size_t)l*4096, b2 + (size_t)l*64, z2, E);
    pack_w3_kernel<<<(4800*64)/256, 256, 0, stream>>>(w3 + (size_t)l*64*4800, w3p);
    zero_kernel<<<((size_t)N*144+255)/256, 256, 0, stream>>>(agg, N*144);
    zero_kernel<<<1, 256, 0, stream>>>(stats, 128);
    fused_tp_kernel<<<E/16, 32, 0, stream>>>(
        z2, w3p, b3 + (size_t)l*4800, h, shb, eidx, Cbuf, agg, E);
    node_update_kernel<<<N, 32, 0, stream>>>(
        agg, si0 + (size_t)l*1024, si1 + (size_t)l*256, si2 + (size_t)l*64,
        tmp, stats, csilu, csig);
    scales_kernel<<<1, 64, 0, stream>>>(stats, bnw + (size_t)l*56, scales, N);
    apply_norm_kernel<<<((size_t)N*120+255)/256, 256, 0, stream>>>(
        tmp, scales, bnb + (size_t)l*32, h, N);
  }

  // --- readout ---
  zero_kernel<<<1, 256, 0, stream>>>(gsum, 528);
  pool_kernel<<<(N+255)/256, 256, 0, stream>>>(h, batch, gsum, N);
  head_kernel<<<1, 16, 0, stream>>>(gsum, hw1, hb1, hw2, hb2, hw3, hb3, (float*)d_out);
}
